// SelfAttentionModule_52716428591531
// MI455X (gfx1250) — matrix operand order repaired and checked by reading, compiled, not run on hardware
//
#include <hip/hip_runtime.h>
#include <hip/hip_bf16.h>
#include <math.h>

typedef __bf16 bhalf;
typedef __attribute__((ext_vector_type(16))) __bf16 v16bf;
typedef __attribute__((ext_vector_type(8)))  __bf16 v8bf;
typedef __attribute__((ext_vector_type(8)))  float   v8f;

#define BB 4
#define CC 256
#define NN 4096

// Scheduling fence: keep batched global loads issued ahead of compute instead of
// letting the scheduler sink them onto their LDS-store uses (serializing latency).
__device__ __forceinline__ void sched_fence() {
  asm volatile("" ::: "memory");
  __builtin_amdgcn_sched_barrier(0);
}

__device__ __forceinline__ v8f wmma_bf16(v16bf a, v16bf b, v8f c) {
  // D = A(16x32 bf16) * B(32x16 bf16) + C(16x16 f32)
  return __builtin_amdgcn_wmma_f32_16x16x32_bf16(false, a, false, b, (short)0, c, false, false);
}

__device__ __forceinline__ v16bf concat8(v8bf lo, v8bf hi) {
  return __builtin_shufflevector(lo, hi, 0, 1, 2, 3, 4, 5, 6, 7, 8, 9, 10, 11, 12, 13, 14, 15);
}

// A fragment (16x32 bf16). A(m,k) = p[m*ld + k], k-contiguous rows; 16B-aligned, ld%8==0.
__device__ __forceinline__ v16bf load_a_frag(const bhalf* p, int ld, int lane) {
  int m = lane & 15, h = lane >> 4;
  const bhalf* r = p + m * ld + h * 8;
  return concat8(*(const v8bf*)r, *(const v8bf*)(r + 16));
}

// B fragment (32x16 bf16). B(k,n) = p[n*ld + k], k-contiguous rows (two 16B runs per lane, same k order as A).
__device__ __forceinline__ v16bf load_b_frag(const bhalf* p, int ld, int lane) {
  int n = lane & 15, h = lane >> 4;
  const bhalf* r = p + n * ld + h * 8;
  return concat8(*(const v8bf*)r, *(const v8bf*)(r + 16));
}

// ---------------- GroupNorm -> bf16 xn, token-major [B,N,C] ----------------
__global__ __launch_bounds__(256) void gn_kernel(const float* __restrict__ x,
                                                 const float* __restrict__ gsc,
                                                 const float* __restrict__ gbi,
                                                 bhalf* __restrict__ xnt) {
  __shared__ float rs[256], rq[256];
  __shared__ __align__(16) bhalf T[256 * 40];   // [n_local][c_local] transpose tile
  int bg = blockIdx.x;                           // (b,g)
  int b = bg >> 3, g = bg & 7;
  const size_t base = ((size_t)b * CC + g * 32) * NN;  // 131072 floats per group
  const float4* xv = (const float4*)(x + base);
  int tid = threadIdx.x;
  float s = 0.f, sq = 0.f;
#pragma unroll 4
  for (int it = 0; it < 128; ++it) {
    float4 v = xv[tid + it * 256];
    s  += v.x + v.y + v.z + v.w;
    sq += v.x * v.x + v.y * v.y + v.z * v.z + v.w * v.w;
  }
  rs[tid] = s; rq[tid] = sq;
  __syncthreads();
  for (int off = 128; off > 0; off >>= 1) {
    if (tid < off) { rs[tid] += rs[tid + off]; rq[tid] += rq[tid + off]; }
    __syncthreads();
  }
  float mu   = rs[0] * (1.f / 131072.f);
  float var  = rq[0] * (1.f / 131072.f) - mu * mu;
  float rstd = rsqrtf(var + 1e-5f);
  int cl = tid >> 3, nc = tid & 7;
  float scv = gsc[g * 32 + cl] * rstd;
  float biv = gbi[g * 32 + cl] - mu * scv;
  for (int nb = 0; nb < 16; ++nb) {
    const float4* src = (const float4*)(x + base + (size_t)cl * NN + nb * 256 + nc * 32);
#pragma unroll
    for (int i = 0; i < 8; ++i) {
      float4 v = src[i];
      int nl = nc * 32 + i * 4;
      T[(nl + 0) * 40 + cl] = (bhalf)(v.x * scv + biv);
      T[(nl + 1) * 40 + cl] = (bhalf)(v.y * scv + biv);
      T[(nl + 2) * 40 + cl] = (bhalf)(v.z * scv + biv);
      T[(nl + 3) * 40 + cl] = (bhalf)(v.w * scv + biv);
    }
    __syncthreads();
    bhalf* drow = xnt + ((size_t)b * NN + nb * 256 + tid) * CC + g * 32;
    uint4* srow = (uint4*)&T[tid * 40];
#pragma unroll
    for (int i = 0; i < 4; ++i) ((uint4*)drow)[i] = srow[i];
    __syncthreads();
  }
}

// ---------------- cast weights: Wc=[Wq*(log2e/16);Wk;Wv] bf16, Wfc bf16, bc f32 ----------------
__global__ __launch_bounds__(256) void castw_kernel(const float* __restrict__ Wq, const float* __restrict__ bq,
                                                    const float* __restrict__ Wk, const float* __restrict__ bk,
                                                    const float* __restrict__ Wv, const float* __restrict__ bv,
                                                    const float* __restrict__ Wf,
                                                    bhalf* __restrict__ Wc, bhalf* __restrict__ Wfc,
                                                    float* __restrict__ bc) {
  int idx = blockIdx.x * 256 + threadIdx.x;
  // Fold C^-0.5 (=1/16) AND log2(e) into the Q path: softmax then uses exp2 directly.
  const float qs = 0.0625f * 1.44269504088896f;
  if (idx < 768 * 256) {
    int r = idx >> 8, c = idx & 255;
    float v;
    if (r < 256)      v = Wq[r * 256 + c] * qs;
    else if (r < 512) v = Wk[(r - 256) * 256 + c];
    else              v = Wv[(r - 512) * 256 + c];
    Wc[idx] = (bhalf)v;
  }
  if (idx < 256 * 256) Wfc[idx] = (bhalf)Wf[idx];
  if (idx < 768) bc[idx] = (idx < 256) ? bq[idx] * qs : (idx < 512 ? bk[idx - 256] : bv[idx - 512]);
}

// ---------------- QKV GEMM -> Q,K token-major [N,C]; V channel-major [C,N] ----------------
// Double-buffered k-loop: next tile's global loads pinned before the WMMAs, stores after.
__global__ __launch_bounds__(256) void qkv_kernel(const bhalf* __restrict__ Wc,
                                                  const float* __restrict__ bc,
                                                  const bhalf* __restrict__ XNt,
                                                  bhalf* __restrict__ Qt, bhalf* __restrict__ Kt,
                                                  bhalf* __restrict__ Vg) {
  __shared__ __align__(16) bhalf smem[4 * 5120];  // 2 x (As 128x40 + Bs 128x40); Outs aliases after loop
  int n0 = blockIdx.x * 128;
  int m0 = blockIdx.y * 128;     // 0..640
  int b  = blockIdx.z;
  int tid = threadIdx.x, lane = tid & 31, wave = tid >> 5;
  int wr = wave & 3, wc = wave >> 2;
  const bhalf* Xb = XNt + (size_t)b * NN * CC;
  v8f acc[2][4] = {};
  uint4 ra[2], rb[2];
  // prologue: stage k0=0 into buffer 0
#pragma unroll
  for (int it = 0; it < 2; ++it) {
    int idx = tid + it * 256, r = idx >> 2, ch = idx & 3;
    ra[it] = *(const uint4*)&Wc[(m0 + r) * CC + ch * 8];
    rb[it] = *(const uint4*)&Xb[(size_t)(n0 + r) * CC + ch * 8];
  }
  sched_fence();
#pragma unroll
  for (int it = 0; it < 2; ++it) {
    int idx = tid + it * 256, r = idx >> 2, ch = idx & 3;
    *(uint4*)&smem[r * 40 + ch * 8] = ra[it];
    *(uint4*)&smem[5120 + r * 40 + ch * 8] = rb[it];
  }
  __syncthreads();
  for (int kk = 0; kk < 8; ++kk) {
    int cur = kk & 1;
    bhalf* Asb = smem + cur * 10240;
    bhalf* Bsb = Asb + 5120;
    if (kk < 7) {
      int k0 = (kk + 1) * 32;
#pragma unroll
      for (int it = 0; it < 2; ++it) {
        int idx = tid + it * 256, r = idx >> 2, ch = idx & 3;
        ra[it] = *(const uint4*)&Wc[(m0 + r) * CC + k0 + ch * 8];
        rb[it] = *(const uint4*)&Xb[(size_t)(n0 + r) * CC + k0 + ch * 8];
      }
      sched_fence();   // loads stay issued here; latency hidden under the WMMAs below
    }
    v16bf af[2];
#pragma unroll
    for (int mi = 0; mi < 2; ++mi) af[mi] = load_a_frag(Asb + (wr * 2 + mi) * 16 * 40, 40, lane);
#pragma unroll
    for (int ni = 0; ni < 4; ++ni) {
      v16bf bfr = load_b_frag(Bsb + (wc * 4 + ni) * 16 * 40, 40, lane);
#pragma unroll
      for (int mi = 0; mi < 2; ++mi) acc[mi][ni] = wmma_bf16(af[mi], bfr, acc[mi][ni]);
    }
    if (kk < 7) {
      bhalf* Asn = smem + (cur ^ 1) * 10240;
#pragma unroll
      for (int it = 0; it < 2; ++it) {
        int idx = tid + it * 256, r = idx >> 2, ch = idx & 3;
        *(uint4*)&Asn[r * 40 + ch * 8] = ra[it];
        *(uint4*)&Asn[5120 + r * 40 + ch * 8] = rb[it];
      }
    }
    __syncthreads();
  }
  // epilogue: bias, bounce through LDS (aliases staging buffers — k-loop fully done)
  bhalf* Outs = smem;   // 128*136 = 17408 elems <= 20480
  int h = lane >> 4, l15 = lane & 15;
  bool isV = (m0 >= 512);
#pragma unroll
  for (int mi = 0; mi < 2; ++mi)
#pragma unroll
    for (int ni = 0; ni < 4; ++ni)
#pragma unroll
      for (int v = 0; v < 8; ++v) {
        int ol = wr * 32 + mi * 16 + v + h * 8;
        int nl = (wc * 4 + ni) * 16 + l15;
        float val = acc[mi][ni][v] + bc[m0 + ol];
        if (isV) Outs[ol * 136 + nl] = (bhalf)val;   // [o][n] for channel-major store
        else     Outs[nl * 136 + ol] = (bhalf)val;   // [n][o] for token-major store
      }
  __syncthreads();
  if (!isV) {
    bhalf* dst = (m0 < 256) ? Qt : Kt;
    int ob = (m0 < 256) ? m0 : (m0 - 256);
    dst += (size_t)b * NN * CC;
#pragma unroll
    for (int it = 0; it < 8; ++it) {
      int idx = tid + it * 256;
      int n = idx >> 4, ch = idx & 15;
      *(uint4*)&dst[(size_t)(n0 + n) * CC + ob + ch * 8] = *(uint4*)&Outs[n * 136 + ch * 8];
    }
  } else {
    bhalf* dst = Vg + (size_t)b * CC * NN;
    int ob = m0 - 512;
#pragma unroll
    for (int it = 0; it < 8; ++it) {
      int idx = tid + it * 256;
      int o = idx >> 4, ch = idx & 15;
      *(uint4*)&dst[(size_t)(ob + o) * NN + n0 + ch * 8] = *(uint4*)&Outs[o * 136 + ch * 8];
    }
  }
}

// ---------------- Flash attention: O[b,i,c] = softmax(Q K^T) V ----------------
#define KLD 264
#define VLD 72
__global__ __launch_bounds__(128) void attn_kernel(const bhalf* __restrict__ Qt,
                                                   const bhalf* __restrict__ Kt,
                                                   const bhalf* __restrict__ Vg,
                                                   bhalf* __restrict__ Ot) {
  __shared__ __align__(16) bhalf KVbuf[256 * VLD];   // K phase [j][c] (KLD) / V phase [c][j] (VLD)
  __shared__ __align__(16) bhalf Ps[4][16 * 72];     // per-wave P tile [m][j]
  int i0 = blockIdx.x * 64;
  int b  = blockIdx.y;
  int tid = threadIdx.x, lane = tid & 31, wave = tid >> 5;
  int h = lane >> 4, l15 = lane & 15;
  const size_t base = (size_t)b * NN * CC;      // Qt/Kt/Ot token-major base
  const size_t bV   = (size_t)b * CC * NN;      // Vg channel-major base
  // Q fragments in registers (1/16 * log2e pre-folded into Wq/bq)
  v16bf qf[8];
  {
    const bhalf* qp = Qt + base + (size_t)(i0 + wave * 16) * CC;
#pragma unroll
    for (int kf = 0; kf < 8; ++kf) qf[kf] = load_a_frag(qp + kf * 32, CC, lane);
  }
  v8f oacc[16] = {};
  float mrun[8], lrun[8];
#pragma unroll
  for (int v = 0; v < 8; ++v) { mrun[v] = -__builtin_inff(); lrun[v] = 0.f; }
  bhalf* Psw = &Ps[wave][0];
  for (int j0 = 0; j0 < NN; j0 += 64) {
    // stage K block [j][c]; 4 global loads in flight per chunk
#pragma unroll
    for (int gch = 0; gch < 4; ++gch) {
      uint4 t[4];
#pragma unroll
      for (int u = 0; u < 4; ++u) {
        int idx = tid + (gch * 4 + u) * 128;
        int j = idx >> 5, ch = idx & 31;
        t[u] = *(const uint4*)&Kt[base + (size_t)(j0 + j) * CC + ch * 8];
      }
      sched_fence();
#pragma unroll
      for (int u = 0; u < 4; ++u) {
        int idx = tid + (gch * 4 + u) * 128;
        int j = idx >> 5, ch = idx & 31;
        *(uint4*)&KVbuf[j * KLD + ch * 8] = t[u];
      }
    }
    // prefetch next K/V block into L2 while we compute
    if (j0 + 64 < NN) {
      int nj = j0 + 64;
      __builtin_prefetch(&Kt[base + (size_t)(nj + (tid >> 1)) * CC + (tid & 1) * 128], 0, 1);
      __builtin_prefetch(&Vg[bV + (size_t)(tid * 2) * NN + nj], 0, 1);
      __builtin_prefetch(&Vg[bV + (size_t)(tid * 2 + 1) * NN + nj], 0, 1);
    }
    __syncthreads();
    // S = Q * K^T  (16 rows x 64 cols per wave) -- already in log2 domain
    v8f s[4] = {};
#pragma unroll
    for (int ni = 0; ni < 4; ++ni)
#pragma unroll
      for (int kf = 0; kf < 8; ++kf) {
        v16bf bfr = load_b_frag(&KVbuf[(ni * 16) * KLD + kf * 32], KLD, lane);
        s[ni] = wmma_bf16(qf[kf], bfr, s[ni]);
      }
    __syncthreads();   // all waves done reading K
    // online softmax in exp2 domain; row (v + h*8) lives across 16 lanes of this half-wave
    float fac[8];
#pragma unroll
    for (int v = 0; v < 8; ++v) {
      float mx = fmaxf(fmaxf(s[0][v], s[1][v]), fmaxf(s[2][v], s[3][v]));
#pragma unroll
      for (int m = 1; m < 16; m <<= 1) mx = fmaxf(mx, __shfl_xor(mx, m, 32));
      float mnew = fmaxf(mrun[v], mx);
      fac[v] = exp2f(mrun[v] - mnew);
      float rsum = 0.f;
#pragma unroll
      for (int ni = 0; ni < 4; ++ni) {
        float p = exp2f(s[ni][v] - mnew);
        s[ni][v] = p;
        rsum += p;
      }
#pragma unroll
      for (int m = 1; m < 16; m <<= 1) rsum += __shfl_xor(rsum, m, 32);
      lrun[v] = lrun[v] * fac[v] + rsum;
      mrun[v] = mnew;
    }
#pragma unroll
    for (int nt = 0; nt < 16; ++nt)
#pragma unroll
      for (int v = 0; v < 8; ++v) oacc[nt][v] *= fac[v];
    // P -> per-wave LDS (C-layout -> A-layout bounce), same-wave RAW (DScnt-ordered)
#pragma unroll
    for (int ni = 0; ni < 4; ++ni)
#pragma unroll
      for (int v = 0; v < 8; ++v)
        Psw[(v + h * 8) * 72 + ni * 16 + l15] = (bhalf)s[ni][v];
    // stage V block [c][j] from channel-major Vg
#pragma unroll
    for (int gch = 0; gch < 4; ++gch) {
      uint4 t[4];
#pragma unroll
      for (int u = 0; u < 4; ++u) {
        int idx = tid + (gch * 4 + u) * 128;
        int c = idx >> 3, ch = idx & 7;
        t[u] = *(const uint4*)&Vg[bV + (size_t)c * NN + j0 + ch * 8];
      }
      sched_fence();
#pragma unroll
      for (int u = 0; u < 4; ++u) {
        int idx = tid + (gch * 4 + u) * 128;
        int c = idx >> 3, ch = idx & 7;
        *(uint4*)&KVbuf[c * VLD + ch * 8] = t[u];
      }
    }
    __syncthreads();
    // O += P * V
#pragma unroll
    for (int ks = 0; ks < 2; ++ks) {
      v16bf af = load_a_frag(Psw + ks * 32, 72, lane);
#pragma unroll
      for (int nt = 0; nt < 16; ++nt) {
        v16bf bfr = load_b_frag(&KVbuf[(nt * 16) * VLD + ks * 32], VLD, lane);
        oacc[nt] = wmma_bf16(af, bfr, oacc[nt]);
      }
    }
    __syncthreads();   // before next K staging
  }
  float rl[8];
#pragma unroll
  for (int v = 0; v < 8; ++v) rl[v] = 1.0f / lrun[v];
  bhalf* op = Ot + base;
#pragma unroll
  for (int nt = 0; nt < 16; ++nt)
#pragma unroll
    for (int v = 0; v < 8; ++v) {
      int irow = i0 + wave * 16 + v + h * 8;
      op[(size_t)irow * CC + nt * 16 + l15] = (bhalf)(oacc[nt][v] * rl[v]);
    }
}

// ---------------- Output projection + bias + residual (double-buffered) ----------------
__global__ __launch_bounds__(256) void proj_kernel(const bhalf* __restrict__ Wfc,
                                                   const float* __restrict__ bfv,
                                                   const bhalf* __restrict__ Ob,
                                                   const float* __restrict__ x,
                                                   float* __restrict__ out) {
  __shared__ __align__(16) bhalf smem[4 * 5120];  // 2 x (As + Bs)
  int n0 = blockIdx.x * 128;
  int m0 = blockIdx.y * 128;
  int b  = blockIdx.z;
  int tid = threadIdx.x, lane = tid & 31, wave = tid >> 5;
  int wr = wave & 3, wc = wave >> 2;
  const bhalf* Obb = Ob + (size_t)b * NN * CC;
  v8f acc[2][4] = {};
  uint4 ra[2], rb[2];
#pragma unroll
  for (int it = 0; it < 2; ++it) {
    int idx = tid + it * 256, r = idx >> 2, ch = idx & 3;
    ra[it] = *(const uint4*)&Wfc[(m0 + r) * CC + ch * 8];
    rb[it] = *(const uint4*)&Obb[(size_t)(n0 + r) * CC + ch * 8];
  }
  sched_fence();
#pragma unroll
  for (int it = 0; it < 2; ++it) {
    int idx = tid + it * 256, r = idx >> 2, ch = idx & 3;
    *(uint4*)&smem[r * 40 + ch * 8] = ra[it];
    *(uint4*)&smem[5120 + r * 40 + ch * 8] = rb[it];
  }
  __syncthreads();
  for (int kk = 0; kk < 8; ++kk) {
    int cur = kk & 1;
    bhalf* Asb = smem + cur * 10240;
    bhalf* Bsb = Asb + 5120;
    if (kk < 7) {
      int k0 = (kk + 1) * 32;
#pragma unroll
      for (int it = 0; it < 2; ++it) {
        int idx = tid + it * 256, r = idx >> 2, ch = idx & 3;
        ra[it] = *(const uint4*)&Wfc[(m0 + r) * CC + k0 + ch * 8];
        rb[it] = *(const uint4*)&Obb[(size_t)(n0 + r) * CC + k0 + ch * 8];
      }
      sched_fence();
    }
    v16bf af[2];
#pragma unroll
    for (int mi = 0; mi < 2; ++mi) af[mi] = load_a_frag(Asb + (wr * 2 + mi) * 16 * 40, 40, lane);
#pragma unroll
    for (int ni = 0; ni < 4; ++ni) {
      v16bf bfr = load_b_frag(Bsb + (wc * 4 + ni) * 16 * 40, 40, lane);
#pragma unroll
      for (int mi = 0; mi < 2; ++mi) acc[mi][ni] = wmma_bf16(af[mi], bfr, acc[mi][ni]);
    }
    if (kk < 7) {
      bhalf* Asn = smem + (cur ^ 1) * 10240;
#pragma unroll
      for (int it = 0; it < 2; ++it) {
        int idx = tid + it * 256, r = idx >> 2, ch = idx & 3;
        *(uint4*)&Asn[r * 40 + ch * 8] = ra[it];
        *(uint4*)&Asn[5120 + r * 40 + ch * 8] = rb[it];
      }
    }
    __syncthreads();
  }
  int h = lane >> 4, l15 = lane & 15;
#pragma unroll
  for (int mi = 0; mi < 2; ++mi)
#pragma unroll
    for (int ni = 0; ni < 4; ++ni)
#pragma unroll
      for (int v = 0; v < 8; ++v) {
        int o = m0 + wr * 32 + mi * 16 + v + h * 8;
        int i = n0 + (wc * 4 + ni) * 16 + l15;
        size_t off = ((size_t)b * CC + o) * NN + i;
        out[off] = acc[mi][ni][v] + bfv[o] + x[off];
      }
}

extern "C" void kernel_launch(void* const* d_in, const int* in_sizes, int n_in,
                              void* d_out, int out_size, void* d_ws, size_t ws_size,
                              hipStream_t stream) {
  const float* x   = (const float*)d_in[0];
  const float* gsc = (const float*)d_in[1];
  const float* gbi = (const float*)d_in[2];
  const float* Wq  = (const float*)d_in[3];
  const float* bq  = (const float*)d_in[4];
  const float* Wk  = (const float*)d_in[5];
  const float* bk  = (const float*)d_in[6];
  const float* Wv  = (const float*)d_in[7];
  const float* bv  = (const float*)d_in[8];
  const float* Wf  = (const float*)d_in[9];
  const float* bfp = (const float*)d_in[10];
  char* ws = (char*)d_ws;
  const size_t SZ = (size_t)BB * CC * NN * sizeof(bhalf);   // 8 MB per bf16 tensor
  bhalf* XNt = (bhalf*)(ws);            // [B,N,C]
  bhalf* Qt  = (bhalf*)(ws + 1 * SZ);   // [B,N,C]
  bhalf* Kt  = (bhalf*)(ws + 2 * SZ);   // [B,N,C]
  bhalf* Vg  = (bhalf*)(ws + 3 * SZ);   // [B,C,N] (channel-major for O-GEMM B tiles)
  bhalf* Ot  = (bhalf*)(ws + 4 * SZ);   // [B,N,C]
  bhalf* Wc  = (bhalf*)(ws + 5 * SZ);                              // [768,256]
  bhalf* Wfc = (bhalf*)(ws + 5 * SZ + 768 * 256 * sizeof(bhalf));  // [256,256]
  float* bc  = (float*)(ws + 5 * SZ + (768 + 256) * 256 * sizeof(bhalf)); // [768]

  gn_kernel<<<32, 256, 0, stream>>>(x, gsc, gbi, XNt);
  castw_kernel<<<1024, 256, 0, stream>>>(Wq, bq, Wk, bk, Wv, bv, Wf, Wc, Wfc, bc);
  qkv_kernel<<<dim3(NN / 128, 6, BB), 256, 0, stream>>>(Wc, bc, XNt, Qt, Kt, Vg);
  attn_kernel<<<dim3(NN / 64, BB), 128, 0, stream>>>(Qt, Kt, Vg, Ot);
  proj_kernel<<<dim3(NN / 128, 2, BB), 256, 0, stream>>>(Wfc, bfp, Ot, x, (float*)d_out);
}
